// DNN_IVA_28879360098438
// MI455X (gfx1250) — matrix operand order repaired and checked by reading, not compiled, not run on hardware
//
#include <hip/hip_runtime.h>
#include <cmath>

typedef __attribute__((ext_vector_type(16))) _Float16 v16h;
typedef __attribute__((ext_vector_type(8)))  _Float16 v8h;
typedef __attribute__((ext_vector_type(8)))  float    v8f;

#define BB   4
#define TT   1000
#define CC   2
#define FF   513
#define UU   256
#define NITER 15
#define EPSV 1e-6f

#define FP    544    // F padded to 17*32 for W1t rows (zero-filled)
#define F2R   640    // F padded to 5*128 rows for W2t (zero-filled)
#define HROWS 1024   // T padded rows for H (t-major, u-contiguous)

#define BM 128
#define BN 64
#define BK 32
#define ASTR  40     // halves; 80B rows -> 16B-aligned chunks, low bank conflict
#define BTSTR 40

// ---------------- CDNA5 async global->LDS copy (ASYNCcnt) -------------------
__device__ __forceinline__ void async_b128(unsigned lds_off, unsigned voff, const void* base) {
  asm volatile("global_load_async_to_lds_b128 %0, %1, %2"
               :: "v"(lds_off), "v"(voff), "s"(base) : "memory");
}
__device__ __forceinline__ void wait_async0() {
  asm volatile("s_wait_asynccnt 0" ::: "memory");
}

// ---------------- fast transcendentals (hardware v_exp/v_log/v_rcp) ---------
__device__ __forceinline__ float fast_feat(float r) {        // log1p(r), r >= 0
  return __logf(1.f + r);
}
__device__ __forceinline__ float fast_tanh(float x) {
  x = fminf(fmaxf(x, -10.f), 10.f);
  float e = __expf(2.f * x);
  return (e - 1.f) * __builtin_amdgcn_rcpf(e + 1.f);
}
__device__ __forceinline__ float fast_sigmoid(float z) {
  return __builtin_amdgcn_rcpf(1.f + __expf(-z));
}

// ---------------- WMMA fragment helpers (v_wmma_f32_16x16x32_f16) -----------
__device__ __forceinline__ v16h frag_a_lds(const _Float16* s, int row0, int stride) {
  int lane = threadIdx.x & 31;
  int m    = lane & 15;
  int kb   = (lane >> 4) << 3;
  const _Float16* p = s + (row0 + m) * stride + kb;
  v8h lo = *(const v8h*)p;
  v8h hi = *(const v8h*)(p + 16);
  return __builtin_shufflevector(lo, hi, 0,1,2,3,4,5,6,7,8,9,10,11,12,13,14,15);
}
__device__ __forceinline__ v16h frag_b_ldsT(const _Float16* s, int col0, int stride) {
  int lane = threadIdx.x & 31;
  int n    = lane & 15;
  int kb   = (lane >> 4) << 3;
  const _Float16* p = s + (col0 + n) * stride + kb;
  v8h lo = *(const v8h*)p;
  v8h hi = *(const v8h*)(p + 16);
  return __builtin_shufflevector(lo, hi, 0,1,2,3,4,5,6,7,8,9,10,11,12,13,14,15);
}

// ---------------- init: (B,T,C,F) -> X (B,C,F,T) complex; ref = X[:,0] ------
__global__ __launch_bounds__(256) void init_X(const float* __restrict__ dr,
                                              const float* __restrict__ di,
                                              float2* __restrict__ X,
                                              float2* __restrict__ ref) {
  size_t i = (size_t)blockIdx.x * 256 + threadIdx.x;
  const size_t total = (size_t)BB * CC * FF * TT;
  if (i >= total) return;
  int t = (int)(i % TT);
  size_t r = i / TT;
  int f = (int)(r % FF); r /= FF;
  int c = (int)(r % CC);
  int b = (int)(r / CC);
  size_t src = ((size_t)(b * TT + t) * CC + c) * FF + f;
  float2 v = make_float2(dr[src], di[src]);
  X[i] = v;
  if (c == 0) ref[((size_t)b * FF + f) * TT + t] = v;
}

// -------- weights -> f16, transposed + padded (zero-filled) layouts ---------
__global__ __launch_bounds__(256) void conv_w(const float* __restrict__ W1,
                                              const float* __restrict__ W2,
                                              _Float16* __restrict__ W1t,
                                              _Float16* __restrict__ W2t) {
  int i = blockIdx.x * 256 + threadIdx.x;
  if (i < UU * FP) {
    int u = i / FP, f = i % FP;
    W1t[i] = (_Float16)(f < FF ? W1[(size_t)f * UU + u] : 0.f);
  }
  if (i < F2R * UU) {
    int f = i / UU, u = i % UU;
    W2t[i] = (_Float16)(f < FF ? W2[(size_t)u * FF + f] : 0.f);
  }
}

// h[t][u] = tanh(W1^T * feat + b1); feat = log1p(|X|^2) fused; double-buffered.
__global__ __launch_bounds__(256) void mask_gemm1(const float2* __restrict__ X,
                                                  const _Float16* __restrict__ W1t,
                                                  const float* __restrict__ b1,
                                                  _Float16* __restrict__ Hh) {
  __shared__ _Float16 As[2][BM * ASTR];
  __shared__ _Float16 Bt[2][BN * BTSTR];
  int bc = blockIdx.z, m0 = blockIdx.y * BM, n0 = blockIdx.x * BN;
  int tid = threadIdx.x, wave = tid >> 5;
  int wm = (wave & 3) * 32, wn = (wave >> 2) * 32;
  const float2* Xbc = X + (size_t)bc * FF * TT;
  int am  = tid >> 1;
  int ak0 = (tid & 1) * 16;
  int bn  = tid & 63;
  int bk  = (tid >> 6) * 8;
  int tcl = n0 + bn; tcl = tcl < TT - 1 ? tcl : TT - 1;     // clamped t for staging
  // stage A (2 async b128) + B (compute, 1 ds_store_b128) into buffer sel
  auto stage = [&](int sel, int k0) {
    unsigned aoff = (unsigned)(size_t)&As[sel][am * ASTR + ak0];
    unsigned voffA = (unsigned)(((m0 + am) * FP + k0 + ak0) * 2);
    async_b128(aoff,      voffA,      W1t);
    async_b128(aoff + 16, voffA + 16, W1t);
    v8h bh;
#pragma unroll
    for (int i = 0; i < 8; ++i) {
      int f = k0 + bk + i;
      f = f < FF - 1 ? f : FF - 1;                           // clamp; pad cols hit zero A
      float2 x = Xbc[(size_t)f * TT + tcl];
      bh[i] = (_Float16)fast_feat(x.x * x.x + x.y * x.y);
    }
    *(v8h*)&Bt[sel][bn * BTSTR + bk] = bh;
  };
  v8f acc00 = {}; v8f acc01 = {}; v8f acc10 = {}; v8f acc11 = {};
  stage(0, 0);
  int buf = 0;
  for (int k0 = 0; k0 < FP; k0 += BK) {
    wait_async0();
    __syncthreads();
    if (k0 + BK < FP) stage(buf ^ 1, k0 + BK);
    v16h a0  = frag_a_lds(As[buf], wm, ASTR);
    v16h a1  = frag_a_lds(As[buf], wm + 16, ASTR);
    v16h bf0 = frag_b_ldsT(Bt[buf], wn, BTSTR);
    v16h bf1 = frag_b_ldsT(Bt[buf], wn + 16, BTSTR);
    acc00 = __builtin_amdgcn_wmma_f32_16x16x32_f16(false, a0, false, bf0, (short)0, acc00, false, false);
    acc01 = __builtin_amdgcn_wmma_f32_16x16x32_f16(false, a0, false, bf1, (short)0, acc01, false, false);
    acc10 = __builtin_amdgcn_wmma_f32_16x16x32_f16(false, a1, false, bf0, (short)0, acc10, false, false);
    acc11 = __builtin_amdgcn_wmma_f32_16x16x32_f16(false, a1, false, bf1, (short)0, acc11, false, false);
    buf ^= 1;
  }
  // epilogue: 8 consecutive u per lane -> one b128 store per sub-tile (t rows padded)
  int lane = tid & 31, col = lane & 15, rbase = (lane >> 4) * 8;
  _Float16* Hbc = Hh + (size_t)bc * HROWS * UU;
  v8f accs[4] = {acc00, acc01, acc10, acc11};
#pragma unroll
  for (int ij = 0; ij < 4; ++ij) {
    int mi = (ij >> 1) * 16, nj = (ij & 1) * 16;
    int u0 = m0 + wm + mi + rbase;
    int t  = n0 + wn + nj + col;                             // < HROWS always
    v8f a = accs[ij];
    v8h hv;
#pragma unroll
    for (int r = 0; r < 8; ++r) hv[r] = (_Float16)fast_tanh(a[r] + b1[u0 + r]);
    *(v8h*)&Hbc[(size_t)t * UU + u0] = hv;
  }
}

// mask[f][t] = sigmoid(W2^T * h + b2); all tiles async copies; double-buffered.
__global__ __launch_bounds__(256) void mask_gemm2(const _Float16* __restrict__ Hh,
                                                  const _Float16* __restrict__ W2t,
                                                  const float* __restrict__ b2,
                                                  float* __restrict__ Wt) {
  __shared__ _Float16 As[2][BM * ASTR];
  __shared__ _Float16 Bt[2][BN * BTSTR];
  int bc = blockIdx.z, m0 = blockIdx.y * BM, n0 = blockIdx.x * BN;
  int tid = threadIdx.x, wave = tid >> 5;
  int wm = (wave & 3) * 32, wn = (wave >> 2) * 32;
  const _Float16* Hbc = Hh + (size_t)bc * HROWS * UU;
  int am  = tid >> 1;
  int ak0 = (tid & 1) * 16;
  int bn  = tid & 63;
  int bk  = (tid >> 6) * 8;
  auto stage = [&](int sel, int k0) {
    unsigned aoff = (unsigned)(size_t)&As[sel][am * ASTR + ak0];
    unsigned voffA = (unsigned)(((m0 + am) * UU + k0 + ak0) * 2);
    async_b128(aoff,      voffA,      W2t);
    async_b128(aoff + 16, voffA + 16, W2t);
    unsigned boff = (unsigned)(size_t)&Bt[sel][bn * BTSTR + bk];
    unsigned voffB = (unsigned)(((n0 + bn) * UU + k0 + bk) * 2);
    async_b128(boff, voffB, Hbc);
  };
  v8f acc00 = {}; v8f acc01 = {}; v8f acc10 = {}; v8f acc11 = {};
  stage(0, 0);
  int buf = 0;
  for (int k0 = 0; k0 < UU; k0 += BK) {
    wait_async0();
    __syncthreads();
    if (k0 + BK < UU) stage(buf ^ 1, k0 + BK);
    v16h a0  = frag_a_lds(As[buf], wm, ASTR);
    v16h a1  = frag_a_lds(As[buf], wm + 16, ASTR);
    v16h bf0 = frag_b_ldsT(Bt[buf], wn, BTSTR);
    v16h bf1 = frag_b_ldsT(Bt[buf], wn + 16, BTSTR);
    acc00 = __builtin_amdgcn_wmma_f32_16x16x32_f16(false, a0, false, bf0, (short)0, acc00, false, false);
    acc01 = __builtin_amdgcn_wmma_f32_16x16x32_f16(false, a0, false, bf1, (short)0, acc01, false, false);
    acc10 = __builtin_amdgcn_wmma_f32_16x16x32_f16(false, a1, false, bf0, (short)0, acc10, false, false);
    acc11 = __builtin_amdgcn_wmma_f32_16x16x32_f16(false, a1, false, bf1, (short)0, acc11, false, false);
    buf ^= 1;
  }
  int lane = tid & 31, col = lane & 15, rbase = (lane >> 4) * 8;
  float* Wbc = Wt + (size_t)bc * FF * TT;
  v8f accs[4] = {acc00, acc01, acc10, acc11};
#pragma unroll
  for (int ij = 0; ij < 4; ++ij) {
    int mi = (ij >> 1) * 16, nj = (ij & 1) * 16;
    v8f a = accs[ij];
#pragma unroll
    for (int r = 0; r < 8; ++r) {
      int f = m0 + wm + mi + rbase + r;
      int t = n0 + wn + nj + col;
      if (f < FF && t < TT)
        Wbc[(size_t)f * TT + t] = fast_sigmoid(a[r] + b2[f]);
    }
  }
}

// ---------------- ISS iteration: one block per (b,f), both sources ----------
__global__ __launch_bounds__(256) void iss_iter(float2* __restrict__ X,
                                                const float* __restrict__ Wt) {
  int b = blockIdx.x / FF;
  int f = blockIdx.x % FF;
  int tid = threadIdx.x;
  __shared__ float wsum[8][6];
  __shared__ float2 vsh[CC];
  float2* X0 = X + ((size_t)(b * CC + 0) * FF + f) * TT;
  float2* X1 = X + ((size_t)(b * CC + 1) * FF + f) * TT;
  const float* W0  = Wt + ((size_t)(b * CC + 0) * FF + f) * TT;
  const float* W1p = Wt + ((size_t)(b * CC + 1) * FF + f) * TT;
  for (int s = 0; s < CC; ++s) {
    float2* Xs = s ? X1 : X0;
    float nr0 = 0.f, ni0 = 0.f, de0 = 0.f, nr1 = 0.f, ni1 = 0.f, de1 = 0.f;
    for (int t = tid; t < TT; t += 256) {
      float2 xs = Xs[t];
      float asq = xs.x * xs.x + xs.y * xs.y;
      float2 x0 = X0[t]; float w0 = W0[t];
      nr0 += w0 * (x0.x * xs.x + x0.y * xs.y);
      ni0 += w0 * (x0.y * xs.x - x0.x * xs.y);
      de0 += w0 * asq;
      float2 x1 = X1[t]; float w1 = W1p[t];
      nr1 += w1 * (x1.x * xs.x + x1.y * xs.y);
      ni1 += w1 * (x1.y * xs.x - x1.x * xs.y);
      de1 += w1 * asq;
    }
#pragma unroll
    for (int off = 16; off > 0; off >>= 1) {
      nr0 += __shfl_down(nr0, off);
      ni0 += __shfl_down(ni0, off);
      de0 += __shfl_down(de0, off);
      nr1 += __shfl_down(nr1, off);
      ni1 += __shfl_down(ni1, off);
      de1 += __shfl_down(de1, off);
    }
    int lane = tid & 31, wv = tid >> 5;
    if (lane == 0) {
      wsum[wv][0] = nr0; wsum[wv][1] = ni0; wsum[wv][2] = de0;
      wsum[wv][3] = nr1; wsum[wv][4] = ni1; wsum[wv][5] = de1;
    }
    __syncthreads();
    if (tid == 0) {
      float a0 = 0.f, a1 = 0.f, a2 = 0.f, a3 = 0.f, a4 = 0.f, a5 = 0.f;
#pragma unroll
      for (int w = 0; w < 8; ++w) {
        a0 += wsum[w][0]; a1 += wsum[w][1]; a2 += wsum[w][2];
        a3 += wsum[w][3]; a4 += wsum[w][4]; a5 += wsum[w][5];
      }
      float num_r[2] = {a0, a3}, num_i[2] = {a1, a4}, den[2] = {a2, a5};
#pragma unroll
      for (int c = 0; c < CC; ++c) {
        if (c == s) {
          float d = den[c] / (float)TT;
          d = d > EPSV ? d : EPSV;
          vsh[c] = make_float2(1.f - rsqrtf(d), 0.f);
        } else {
          float d = den[c] > EPSV ? den[c] : EPSV;
          vsh[c] = make_float2(num_r[c] / d, num_i[c] / d);
        }
      }
    }
    __syncthreads();
    float2 v0 = vsh[0], v1 = vsh[1];
    for (int t = tid; t < TT; t += 256) {
      float2 xs = Xs[t];
      float2 x0 = X0[t];
      x0.x -= v0.x * xs.x - v0.y * xs.y;
      x0.y -= v0.x * xs.y + v0.y * xs.x;
      X0[t] = x0;
      float2 x1 = X1[t];
      x1.x -= v1.x * xs.x - v1.y * xs.y;
      x1.y -= v1.x * xs.y + v1.y * xs.x;
      X1[t] = x1;
    }
    __syncthreads();
  }
}

// ---------------- projection back + output transpose (C,B,T,F) --------------
__global__ __launch_bounds__(256) void proj_back(const float2* __restrict__ X,
                                                 const float2* __restrict__ ref,
                                                 float2* __restrict__ out) {
  int b = blockIdx.x / FF;
  int f = blockIdx.x % FF;
  int tid = threadIdx.x;
  __shared__ float wsum[8][6];
  __shared__ float2 csh[CC];
  const float2* X0 = X + ((size_t)(b * CC + 0) * FF + f) * TT;
  const float2* X1 = X + ((size_t)(b * CC + 1) * FF + f) * TT;
  const float2* R  = ref + ((size_t)b * FF + f) * TT;
  float nr0 = 0.f, ni0 = 0.f, de0 = 0.f, nr1 = 0.f, ni1 = 0.f, de1 = 0.f;
  for (int t = tid; t < TT; t += 256) {
    float2 rr = R[t];
    float2 y0 = X0[t];
    nr0 += y0.x * rr.x + y0.y * rr.y;
    ni0 += y0.x * rr.y - y0.y * rr.x;
    de0 += y0.x * y0.x + y0.y * y0.y;
    float2 y1 = X1[t];
    nr1 += y1.x * rr.x + y1.y * rr.y;
    ni1 += y1.x * rr.y - y1.y * rr.x;
    de1 += y1.x * y1.x + y1.y * y1.y;
  }
#pragma unroll
  for (int off = 16; off > 0; off >>= 1) {
    nr0 += __shfl_down(nr0, off);
    ni0 += __shfl_down(ni0, off);
    de0 += __shfl_down(de0, off);
    nr1 += __shfl_down(nr1, off);
    ni1 += __shfl_down(ni1, off);
    de1 += __shfl_down(de1, off);
  }
  int lane = tid & 31, wv = tid >> 5;
  if (lane == 0) {
    wsum[wv][0] = nr0; wsum[wv][1] = ni0; wsum[wv][2] = de0;
    wsum[wv][3] = nr1; wsum[wv][4] = ni1; wsum[wv][5] = de1;
  }
  __syncthreads();
  if (tid == 0) {
    float a0 = 0.f, a1 = 0.f, a2 = 0.f, a3 = 0.f, a4 = 0.f, a5 = 0.f;
#pragma unroll
    for (int w = 0; w < 8; ++w) {
      a0 += wsum[w][0]; a1 += wsum[w][1]; a2 += wsum[w][2];
      a3 += wsum[w][3]; a4 += wsum[w][4]; a5 += wsum[w][5];
    }
    float d0 = a2 > EPSV ? a2 : EPSV;
    float d1 = a5 > EPSV ? a5 : EPSV;
    csh[0] = make_float2(a0 / d0, a1 / d0);
    csh[1] = make_float2(a3 / d1, a4 / d1);
  }
  __syncthreads();
  float2 c0 = csh[0], c1 = csh[1];
  for (int t = tid; t < TT; t += 256) {
    float2 y0 = X0[t];
    out[((size_t)(0 * BB + b) * TT + t) * FF + f] =
        make_float2(y0.x * c0.x - y0.y * c0.y, y0.x * c0.y + y0.y * c0.x);
    float2 y1 = X1[t];
    out[((size_t)(1 * BB + b) * TT + t) * FF + f] =
        make_float2(y1.x * c1.x - y1.y * c1.y, y1.x * c1.y + y1.y * c1.x);
  }
}

// ---------------- launch -----------------------------------------------------
extern "C" void kernel_launch(void* const* d_in, const int* in_sizes, int n_in,
                              void* d_out, int out_size, void* d_ws, size_t ws_size,
                              hipStream_t stream) {
  const float* dr = (const float*)d_in[0];
  const float* di = (const float*)d_in[1];
  // d_in[2] = ilens (unused; all T)
  const float* W1 = (const float*)d_in[3];
  const float* b1 = (const float*)d_in[4];
  const float* W2 = (const float*)d_in[5];
  const float* b2 = (const float*)d_in[6];

  const size_t nX   = (size_t)BB * CC * FF * TT;   // complex
  const size_t nRef = (size_t)BB * FF * TT;        // complex
  float2*   X    = (float2*)d_ws;
  float2*   ref  = X + nX;
  float*    Wt   = (float*)(ref + nRef);           // mask, f32
  _Float16* Hh   = (_Float16*)(Wt + nX);           // hidden (bc,t,u), f16
  _Float16* W1t  = Hh + (size_t)BB * CC * HROWS * UU;
  _Float16* W2t  = W1t + (size_t)UU * FP;

  init_X<<<(unsigned)((nX + 255) / 256), 256, 0, stream>>>(dr, di, X, ref);
  conv_w<<<(F2R * UU + 255) / 256, 256, 0, stream>>>(W1, W2, W1t, W2t);

  dim3 g1((TT + BN - 1) / BN, UU / BM, BB * CC);            // 16 x 2 x 8
  dim3 g2((TT + BN - 1) / BN, (FF + BM - 1) / BM, BB * CC); // 16 x 5 x 8
  for (int it = 0; it < NITER; ++it) {
    mask_gemm1<<<g1, 256, 0, stream>>>(X, W1t, b1, Hh);
    mask_gemm2<<<g2, 256, 0, stream>>>(Hh, W2t, b2, Wt);
    iss_iter<<<BB * FF, 256, 0, stream>>>(X, Wt);
  }
  proj_back<<<BB * FF, 256, 0, stream>>>(X, ref, (float2*)d_out);
}
